// KANExpert_4784593568351
// MI455X (gfx1250) — compile-verified, matrix operand order checked by reading, not run on hardware
//
#include <hip/hip_runtime.h>
#include <hip/hip_bf16.h>

typedef __attribute__((ext_vector_type(16))) __bf16 v16bf;
typedef __attribute__((ext_vector_type(8)))  float  v8f;

#define BM 128
#define BN 128
#define BK 32
#define LROW 80                   // LDS bytes per tile row: 64 data + 16 pad (16B aligned, bank-spread)
#define TILE_BYTES (BM * LROW)    // 10240
#define BUF_BYTES (2 * TILE_BYTES)
#define NSTAGE 3                  // triple buffer: 60KB LDS, 2 K-steps of DMA in flight

// ---------- helpers ----------
__device__ __forceinline__ unsigned short f2bf(float f) {
  unsigned int u = __float_as_uint(f);
  u += 0x7FFFu + ((u >> 16) & 1u);          // round-to-nearest-even
  return (unsigned short)(u >> 16);
}

// Cubic B-spline basis on the fixed uniform grid t_j = 0.4*j - 2.2, j=0..11
__device__ __forceinline__ void spline8(float x, float b[8]) {
  float bb[11];
#pragma unroll
  for (int j = 0; j < 11; ++j) {
    float t0 = 0.4f * (float)j - 2.2f;
    bb[j] = (x >= t0 && x < t0 + 0.4f) ? 1.0f : 0.0f;
  }
#pragma unroll
  for (int k = 1; k <= 3; ++k) {
    float inv = 1.0f / (0.4f * (float)k);
#pragma unroll
    for (int j = 0; j + k < 11; ++j) {
      float tj  = 0.4f * (float)j - 2.2f;
      float tk1 = 0.4f * (float)(j + k + 1) - 2.2f;
      bb[j] = (x - tj) * inv * bb[j] + (tk1 - x) * inv * bb[j + 1];
    }
  }
#pragma unroll
  for (int j = 0; j < 8; ++j) b[j] = bb[j];
}

// CDNA5 async DMA: copy 16B per lane from global directly into LDS (ASYNCcnt-tracked)
__device__ __forceinline__ void async_b128(void* lds_ptr, const void* gptr) {
  unsigned int lds_off = (unsigned int)(size_t)lds_ptr;   // LDS aperture: addr[31:0] = LDS offset
  asm volatile("global_load_async_to_lds_b128 %0, %1, off"
               :: "v"(lds_off), "v"((unsigned long long)(size_t)gptr)
               : "memory");
}

// ---------- expansion kernels ----------
__global__ void k_zero(unsigned int* __restrict__ p, size_t n) {
  size_t i = (size_t)blockIdx.x * blockDim.x + threadIdx.x;
  size_t stride = (size_t)gridDim.x * blockDim.x;
  for (; i < n; i += stride) p[i] = 0u;
}

// A_exp[m, c*in + i] : c==0 -> SiLU(x), c=1..8 -> B-spline basis (bf16)
__global__ void k_expand_act(const float* __restrict__ X,
                             unsigned short* __restrict__ A,
                             unsigned int M, unsigned int in, unsigned int Kpad) {
  unsigned int idx = blockIdx.x * blockDim.x + threadIdx.x;
  if (idx >= M * in) return;
  unsigned int m = idx / in, i = idx - m * in;
  float x = X[idx];
  unsigned short* row = A + (size_t)m * Kpad;
  row[i] = f2bf(x / (1.0f + __expf(-x)));
  float b[8];
  spline8(x, b);
#pragma unroll
  for (int c = 0; c < 8; ++c) row[(size_t)(c + 1) * in + i] = f2bf(b[c]);
}

// W_exp[o, c*in + i] : c==0 -> base_w, c=1..8 -> spline_w*scaler (bf16)
__global__ void k_expand_w(const float* __restrict__ bw,
                           const float* __restrict__ sw,
                           const float* __restrict__ sc,
                           unsigned short* __restrict__ W,
                           unsigned int out, unsigned int in, unsigned int Kpad) {
  unsigned int idx = blockIdx.x * blockDim.x + threadIdx.x;
  if (idx >= out * in) return;
  unsigned int o = idx / in, i = idx - o * in;
  unsigned short* row = W + (size_t)o * Kpad;
  row[i] = f2bf(bw[idx]);
  float s = sc[idx];
  const float* sp = sw + (size_t)idx * 8;
#pragma unroll
  for (int c = 0; c < 8; ++c) row[(size_t)(c + 1) * in + i] = f2bf(sp[c] * s);
}

// ---------- WMMA GEMM: C[M,Nout] = A[M,K] * W[Npad,K]^T (both bf16 row-major, K contiguous) ----------
__global__ __launch_bounds__(256) void k_kan_gemm(
    const unsigned short* __restrict__ A, const unsigned short* __restrict__ W,
    float* __restrict__ C, int K, int Nout, int ldc) {
  __shared__ __align__(16) char lds[NSTAGE * BUF_BYTES];   // 61440 bytes

  const int tid   = threadIdx.x;
  const int lane  = tid & 31;
  const int wave  = tid >> 5;
  const int laneL = lane & 15;
  const int laneH = lane >> 4;
  const int wm    = wave >> 2;   // 0..1 -> 64-row slab in M
  const int wn    = wave & 3;    // 0..3 -> 32-col slab in N
  const int m0    = blockIdx.y * BM;
  const int n0    = blockIdx.x * BN;

  // ---- staging geometry: 256 threads x 4 async b128 = 128x32 bf16 A-tile + B-tile ----
  const int r0 = tid >> 2;            // 0..63 (and r0+64)
  const int pb = (tid & 3) << 4;      // byte part 0/16/32/48 within 64B row
  const char* gA0 = (const char*)A + ((size_t)(m0 + r0)       * K) * 2 + pb;
  const char* gA1 = (const char*)A + ((size_t)(m0 + r0 + 64)  * K) * 2 + pb;
  const char* gB0 = (const char*)W + ((size_t)(n0 + r0)       * K) * 2 + pb;
  const char* gB1 = (const char*)W + ((size_t)(n0 + r0 + 64)  * K) * 2 + pb;
  const int oA0 = r0 * LROW + pb;
  const int oA1 = (r0 + 64) * LROW + pb;
  const int oB0 = TILE_BYTES + oA0;
  const int oB1 = TILE_BYTES + oA1;

  const int nK = K / BK;

  // issue the 4 async DMA ops of K-step kk into LDS buffer (kk % NSTAGE)
  auto stage = [&](int kk) {
    char* db = lds + (kk % NSTAGE) * BUF_BYTES;
    const size_t go = (size_t)kk * (BK * 2);
    async_b128(db + oA0, gA0 + go);
    async_b128(db + oA1, gA1 + go);
    async_b128(db + oB0, gB0 + go);
    async_b128(db + oB1, gB1 + go);
  };

  // prologue: fill the pipeline with NSTAGE-1 stages
#pragma unroll
  for (int s = 0; s < NSTAGE - 1; ++s)
    if (s < nK) stage(s);

  v8f acc[4][2];
#pragma unroll
  for (int i = 0; i < 4; ++i)
#pragma unroll
    for (int j = 0; j < 2; ++j)
#pragma unroll
      for (int e = 0; e < 8; ++e) acc[i][j][e] = 0.0f;

  union BFrag { uint4 u[2]; v16bf v; };
  // fragment: laneL = row within 16-row subtile; laneH picks K-halves {0..7,16..23} vs {8..15,24..31}
  auto load_frag = [&](const char* base, int row) -> v16bf {
    BFrag f;
    const char* p = base + row * LROW + laneH * 16;
    f.u[0] = *(const uint4*)p;
    f.u[1] = *(const uint4*)(p + 32);
    return f.v;
  };

  for (int kk = 0; kk < nK; ++kk) {
    // steady state: stages kk,kk+1 outstanding (8 ops); <=4 left ==> stage kk landed (in-order)
    if (kk + 1 < nK)
      asm volatile("s_wait_asynccnt 4" ::: "memory");
    else
      asm volatile("s_wait_asynccnt 0" ::: "memory");
    __syncthreads();  // stage-kk data visible to all waves; prior reads of buf (kk-1)%3 done

    if (kk + NSTAGE - 1 < nK) stage(kk + NSTAGE - 1);   // refill into buf (kk-1)%3

    const char* la = lds + (kk % NSTAGE) * BUF_BYTES;
    const char* lb = la + TILE_BYTES;
    v16bf af[4], bfr[2];
#pragma unroll
    for (int i = 0; i < 4; ++i) af[i]  = load_frag(la, wm * 64 + i * 16 + laneL);
#pragma unroll
    for (int j = 0; j < 2; ++j) bfr[j] = load_frag(lb, wn * 32 + j * 16 + laneL);

#pragma unroll
    for (int i = 0; i < 4; ++i)
#pragma unroll
      for (int j = 0; j < 2; ++j)
        acc[i][j] = __builtin_amdgcn_wmma_f32_16x16x32_bf16(
            false, af[i], false, bfr[j], (short)0, acc[i][j], false, false);
  }

  // C/D layout: VGPR r, lanes 0-15 -> (M=r, N=lane); lanes 16-31 -> (M=r+8, N=lane-16)
#pragma unroll
  for (int i = 0; i < 4; ++i) {
    const int gm = m0 + wm * 64 + i * 16 + laneH * 8;
#pragma unroll
    for (int j = 0; j < 2; ++j) {
      const int gn = n0 + wn * 32 + j * 16 + laneL;
      if (gn < Nout) {
        float* cp = C + (size_t)gm * ldc + gn;
#pragma unroll
        for (int r = 0; r < 8; ++r) cp[(size_t)r * ldc] = acc[i][j][r];
      }
    }
  }
}

// ---------- host ----------
extern "C" void kernel_launch(void* const* d_in, const int* in_sizes, int n_in,
                              void* d_out, int out_size, void* d_ws, size_t ws_size,
                              hipStream_t stream) {
  (void)in_sizes; (void)n_in; (void)out_size; (void)ws_size;
  const float* x   = (const float*)d_in[0];
  const float* bw1 = (const float*)d_in[1];
  const float* sw1 = (const float*)d_in[2];
  const float* sc1 = (const float*)d_in[3];
  const float* bw2 = (const float*)d_in[4];
  const float* sw2 = (const float*)d_in[5];
  const float* sc2 = (const float*)d_in[6];
  float* out = (float*)d_out;

  const int M = 4096, IN1 = 512, OUT1 = 2049, IN2 = 2049, OUT2 = 512;
  const int K1  = IN1 * 9;                    // 4608 (multiple of 32)
  const int K2  = (IN2 * 9 + 31) & ~31;       // 18464
  const int NP1 = (OUT1 + 127) & ~127;        // 2176
  const int NP2 = OUT2;                       // 512

  char* ws = (char*)d_ws;
  unsigned short* Wx = (unsigned short*)ws;                     // max 20.1 MB (either layer)
  size_t wbytes = (size_t)NP1 * K1 * 2;
  float* H = (float*)(ws + wbytes);                             // 4096x2049 f32, 33.6 MB
  size_t hbytes = (size_t)M * OUT1 * sizeof(float);
  unsigned short* Ax = (unsigned short*)(ws + wbytes + hbytes); // max 4096xK2 bf16, 151 MB

  // ---- layer 1 ----
  k_zero<<<2048, 256, 0, stream>>>((unsigned int*)Wx, (size_t)NP1 * K1 / 2);
  {
    unsigned int n = (unsigned int)OUT1 * IN1;
    k_expand_w<<<(n + 255) / 256, 256, 0, stream>>>(bw1, sw1, sc1, Wx, OUT1, IN1, K1);
  }
  {
    unsigned int n = (unsigned int)M * IN1;
    k_expand_act<<<(n + 255) / 256, 256, 0, stream>>>(x, Ax, M, IN1, K1);
  }
  {
    dim3 g(NP1 / BN, M / BM);
    k_kan_gemm<<<g, 256, 0, stream>>>(Ax, Wx, H, K1, OUT1, OUT1);
  }

  // ---- layer 2 ----
  k_zero<<<2048, 256, 0, stream>>>((unsigned int*)Wx, (size_t)NP2 * K2 / 2);
  {
    unsigned int n = (unsigned int)OUT2 * IN2;
    k_expand_w<<<(n + 255) / 256, 256, 0, stream>>>(bw2, sw2, sc2, Wx, OUT2, IN2, K2);
  }
  {
    unsigned int n = (unsigned int)M * IN2;
    k_expand_act<<<(n + 255) / 256, 256, 0, stream>>>(H, Ax, M, IN2, K2);
  }
  {
    dim3 g(NP2 / BN, M / BM);
    k_kan_gemm<<<g, 256, 0, stream>>>(Ax, Wx, out, K2, OUT2, OUT2);
  }
}
